// CEM_23476291240528
// MI455X (gfx1250) — matrix operand order repaired and checked by reading, not compiled, not run on hardware
//
#include <hip/hip_runtime.h>
#include <math.h>

// ---------------- problem constants (from reference) ----------------
#define PN    4096      // particles
#define TN    128       // horizon
#define AN    32        // action dim
#define KSEL  409       // int(P * 0.1)
#define KPAD  416       // KSEL padded to multiple of 32 (WMMA K-chunk)
#define TEMP  0.5f
#define MOM   0.1f
#define MIN_STD 0.05f
#define FINF  1e30f

typedef __attribute__((ext_vector_type(16))) _Float16 v16h;
typedef __attribute__((ext_vector_type(8)))  float    v8f;
typedef __attribute__((ext_vector_type(4)))  int      v4i;

// ---------------- CDNA5 async global->LDS path (probed) ----------------
#if defined(__has_builtin)
#if __has_builtin(__builtin_amdgcn_global_load_async_to_lds_b128)
#define HAVE_ASYNC_LDS 1
#endif
#endif
#ifndef HAVE_ASYNC_LDS
#define HAVE_ASYNC_LDS 0
#endif

#if HAVE_ASYNC_LDS
typedef __attribute__((address_space(1))) v4i* gv4p;   // global v4i*
typedef __attribute__((address_space(3))) v4i* lv4p;   // LDS v4i*
#if __has_builtin(__builtin_amdgcn_s_wait_asynccnt)
#define WAIT_ASYNC(n) __builtin_amdgcn_s_wait_asynccnt(n)
#else
#define WAIT_ASYNC(n) asm volatile("s_wait_asynccnt %0" :: "n"(n) : "memory")
#endif
#define WAIT_DS0() asm volatile("s_wait_dscnt 0x0" ::: "memory")
#else
#define WAIT_ASYNC(n) do {} while (0)
#define WAIT_DS0()    do {} while (0)
#endif

#define CSTRIDE 132   // 128 + 4 pad floats: keeps B128 16B alignment, breaks bank aliasing

// Stage one 512B cost row (128 floats) for this lane into its LDS column.
__device__ __forceinline__ void stage_row(const float* __restrict__ g, float* l) {
#if HAVE_ASYNC_LDS
    #pragma unroll
    for (int q = 0; q < TN / 4; ++q) {
        __builtin_amdgcn_global_load_async_to_lds_b128(
            (gv4p)(g + q * 4), (lv4p)(l + q * 4), 0, 0);
    }
#else
    #pragma unroll 8
    for (int q = 0; q < TN / 4; ++q)
        ((float4*)l)[q] = ((const float4*)g)[q];
#endif
}

// =====================================================================
// Kernel 1: DTW min-plus DP, one lane per particle, one wave per block.
// DP row lives entirely in VGPRs (128 regs/lane, fully unrolled j-loop)
// -> dependency chain per cell is just min,min,add. Cost rows are
// double-buffered into LDS via GLOBAL_LOAD_ASYNC_TO_LDS_B128 (ASYNCcnt),
// row i+1 in flight while row i computes. 256 MB streamed ~ HBM-bound.
// =====================================================================
__global__ void __launch_bounds__(32) dtw_kernel(const float* __restrict__ obs,
                                                 float* __restrict__ dists) {
    __shared__ __align__(16) float cbuf[2][32 * CSTRIDE];   // 2 x 16.5 KB
    const int tid = threadIdx.x;
    const int p   = blockIdx.x * 32 + tid;
    const float* cp = obs + (size_t)p * (TN * TN);
    float* l0 = &cbuf[0][tid * CSTRIDE];
    float* l1 = &cbuf[1][tid * CSTRIDE];

    stage_row(cp, l0);                    // row 0 in flight

    float row[TN];                        // prev-row D values, register resident
    #pragma unroll
    for (int j = 0; j < TN; ++j) row[j] = FINF;

    float carry = FINF;
    for (int i = 0; i < TN; ++i) {
        if (i + 1 < TN) {
            WAIT_DS0();                   // prior DS reads done before async rewrites buffer
            stage_row(cp + (size_t)(i + 1) * TN, ((i + 1) & 1) ? l1 : l0);
            WAIT_ASYNC(TN / 4);           // <=32 outstanding -> row i landed in LDS
        } else {
            WAIT_ASYNC(0);
        }
        const float* c = (i & 1) ? l1 : l0;
        float diag = (i == 0) ? 0.0f : FINF;   // prev_row[0]
        carry = FINF;
        #pragma unroll
        for (int j = 0; j < TN; ++j) {
            const float up = row[j];                       // prev_row[j+1]
            const float nv = fminf(fminf(up, diag), carry) + c[j];
            row[j] = nv; diag = up; carry = nv;
        }
    }
    dists[p] = carry;   // D[T, T]
}

// =====================================================================
// Kernel 2: deterministic top-K select (single block, 1024 threads).
// Radix binary-search on float bit patterns (all dists >= 0) finds the
// K-th smallest value; prefix-scan compaction (no float atomics ->
// bit-identical every replay) builds elite idx + exp-weights, ascending
// particle index for tie-break (matches lax.top_k).
// =====================================================================
__global__ void __launch_bounds__(1024) select_kernel(const float* __restrict__ dists,
                                                      int*   __restrict__ eidx,
                                                      float* __restrict__ ew,
                                                      float* __restrict__ scal) {
    __shared__ float    sd[PN];        // 16 KB
    __shared__ int      red[1024];
    __shared__ float    fred[1024];
    __shared__ int      eq[64];
    __shared__ float    sew[KPAD];

    const int tid = threadIdx.x;
    for (int i = tid; i < PN; i += 1024) sd[i] = dists[i];
    __syncthreads();

    // ---- global min (== min over elites) ----
    float lm = FINF;
    #pragma unroll
    for (int j = 0; j < 4; ++j) lm = fminf(lm, sd[tid * 4 + j]);
    fred[tid] = lm;
    __syncthreads();
    for (int s = 512; s > 0; s >>= 1) {
        if (tid < s) fred[tid] = fminf(fred[tid], fred[tid + s]);
        __syncthreads();
    }
    const float dmin = fred[0];
    __syncthreads();

    // ---- K-th smallest via 32-step radix search on uint bits ----
    unsigned X = 0;
    for (int b = 31; b >= 0; --b) {
        const unsigned trial = X | (1u << b);
        int c = 0;
        #pragma unroll
        for (int j = 0; j < 4; ++j)
            c += (__float_as_uint(sd[tid * 4 + j]) < trial) ? 1 : 0;
        red[tid] = c;
        __syncthreads();
        for (int s = 512; s > 0; s >>= 1) {
            if (tid < s) red[tid] += red[tid + s];
            __syncthreads();
        }
        if (red[0] < KSEL) X = trial;   // downward-closed -> greedy max = kth value
        __syncthreads();
    }

    // ---- deterministic compaction: strict-less and equal sets ----
    unsigned ub[4];
    int cl = 0, ce = 0;
    #pragma unroll
    for (int j = 0; j < 4; ++j) {
        ub[j] = __float_as_uint(sd[tid * 4 + j]);
        cl += (ub[j] < X) ? 1 : 0;
        ce += (ub[j] == X) ? 1 : 0;
    }
    // inclusive scan of cl
    red[tid] = cl; __syncthreads();
    for (int s = 1; s < 1024; s <<= 1) {
        int v = (tid >= s) ? red[tid - s] : 0;
        __syncthreads();
        red[tid] += v;
        __syncthreads();
    }
    const int baseL = red[tid] - cl;
    const int nLess = red[1023];
    __syncthreads();
    // inclusive scan of ce
    red[tid] = ce; __syncthreads();
    for (int s = 1; s < 1024; s <<= 1) {
        int v = (tid >= s) ? red[tid - s] : 0;
        __syncthreads();
        red[tid] += v;
        __syncthreads();
    }
    const int baseE = red[tid] - ce;
    __syncthreads();

    int wl = 0, we = 0;
    #pragma unroll
    for (int j = 0; j < 4; ++j) {
        const int p = tid * 4 + j;
        if (ub[j] < X) {
            const int pos = baseL + (wl++);
            eidx[pos] = p;
            sew[pos]  = __expf(TEMP * (dmin - sd[p]));
        } else if (ub[j] == X) {
            const int pos = baseE + (we++);
            if (pos < 64) eq[pos] = p;     // ascending particle index
        }
    }
    __syncthreads();

    const int E = KSEL - nLess;            // ties needed at boundary (usually 1)
    if (tid < E && tid < 64) {
        eidx[nLess + tid] = eq[tid];
        sew[nLess + tid]  = __expf(TEMP * (dmin - __uint_as_float(X)));
    }
    if (tid >= KSEL && tid < KPAD) {       // zero-weight padding for WMMA chunks
        sew[tid]  = 0.0f;
        eidx[tid] = 0;
    }
    __syncthreads();

    // ---- W = sum of weights (deterministic tree) ----
    fred[tid] = (tid < KPAD) ? sew[tid] : 0.0f;
    __syncthreads();
    for (int s = 512; s > 0; s >>= 1) {
        if (tid < s) fred[tid] += fred[tid + s];
        __syncthreads();
    }
    if (tid == 0) { scal[0] = fred[0]; scal[1] = dmin; }
    if (tid < KPAD) ew[tid] = sew[tid];
}

// =====================================================================
// Kernel 3: weighted mean/std over elites via V_WMMA_F32_16X16X32_F16.
// One wave per t. M = a-index (two 16-wide tiles for A=32), WMMA-K =
// elites (13 chunks of 32), B = weight vector broadcast across N.
// Two accumulator chains: sum(w*a) and sum(w*a^2). Actions recomputed
// on the fly: clip(mean + std*noise, -1, 1). VGPR layouts per ISA 7.12.2.
// =====================================================================
__global__ void __launch_bounds__(32) cem_reduce_kernel(const float* __restrict__ means,
                                                        const float* __restrict__ stds,
                                                        const float* __restrict__ noise,
                                                        const int*   __restrict__ eidx,
                                                        const float* __restrict__ ew,
                                                        const float* __restrict__ scal,
                                                        float* __restrict__ out) {
    const int t    = blockIdx.x;
    const int lane = threadIdx.x;
    const bool hi  = lane >= 16;
    const int  m   = lane & 15;
    const int  a0  = m;          // tile 0: a in [0,16)
    const int  a1  = 16 + m;     // tile 1: a in [16,32)

    const float mu0 = means[t * AN + a0], sg0 = stds[t * AN + a0];
    const float mu1 = means[t * AN + a1], sg1 = stds[t * AN + a1];
    const float* nt = noise + (size_t)t * (PN * AN);

    // A-operand element -> WMMA K mapping for this lane
    int kA[16];
    #pragma unroll
    for (int e = 0; e < 16; ++e)
        kA[e] = ((e < 8) ? e : e + 8) + (hi ? 8 : 0);
    const int kBoff = hi ? 16 : 0;

    v8f accM0 = {}, accV0 = {}, accM1 = {}, accV1 = {};

    for (int cc = 0; cc < KPAD / 32; ++cc) {
        const int base = cc * 32;
        v16h aH0, aS0, aH1, aS1, wH;
        #pragma unroll
        for (int e = 0; e < 16; ++e) {
            const int p = eidx[base + kA[e]];
            const float n0 = nt[(size_t)p * AN + a0];
            const float n1 = nt[(size_t)p * AN + a1];
            const float x0 = fminf(fmaxf(mu0 + sg0 * n0, -1.0f), 1.0f);
            const float x1 = fminf(fmaxf(mu1 + sg1 * n1, -1.0f), 1.0f);
            aH0[e] = (_Float16)x0;  aS0[e] = (_Float16)(x0 * x0);
            aH1[e] = (_Float16)x1;  aS1[e] = (_Float16)(x1 * x1);
            wH[e]  = (_Float16)ew[base + kA[e]];   // weight for this lane's K
        }
        accM0 = __builtin_amdgcn_wmma_f32_16x16x32_f16(false, aH0, false, wH,
                                                       (short)0, accM0, false, false);
        accV0 = __builtin_amdgcn_wmma_f32_16x16x32_f16(false, aS0, false, wH,
                                                       (short)0, accV0, false, false);
        accM1 = __builtin_amdgcn_wmma_f32_16x16x32_f16(false, aH1, false, wH,
                                                       (short)0, accM1, false, false);
        accV1 = __builtin_amdgcn_wmma_f32_16x16x32_f16(false, aS1, false, wH,
                                                       (short)0, accV1, false, false);
    }

    const float W = scal[0];
    // Every column N holds the same result (B broadcast); lanes 0 and 16
    // cover M = 0..7 and 8..15 respectively.
    if (lane == 0 || lane == 16) {
        const int rb = hi ? 8 : 0;
        #pragma unroll
        for (int r = 0; r < 8; ++r) {
            #pragma unroll
            for (int tile = 0; tile < 2; ++tile) {
                const int  a  = tile * 16 + rb + r;
                const float S1 = (tile == 0) ? accM0[r] : accM1[r];
                const float S2 = (tile == 0) ? accV0[r] : accV1[r];
                const float q1   = S1 / W;                       // sum of normalized w*a
                const float q2   = S2 / W;                       // sum of normalized w*a^2
                const float mean = q1 / (1.0f + 1e-9f);
                float var  = (q2 - 2.0f * mean * q1 + mean * mean) / (1.0f + 1e-9f);
                float sd   = sqrtf(fmaxf(var, 0.0f));
                sd = fminf(fmaxf(sd, MIN_STD), 1.0f);
                const float mnew = MOM * means[t * AN + a] + (1.0f - MOM) * mean;
                out[t * AN + a]            = mnew;   // stack[0] = means_new
                out[TN * AN + t * AN + a]  = sd;     // stack[1] = _std
            }
        }
    }
}

// =====================================================================
// Launch: obs_diffs[P,T,T], means[T,1,A], stds[T,1,A], noise[T,P,A]
// out: [2, T, 1, A] = 8192 floats. ws: dists(4096f) | eidx(416i) |
// ew(416f) | scal(2f)  ~= 20 KB.
// =====================================================================
extern "C" void kernel_launch(void* const* d_in, const int* in_sizes, int n_in,
                              void* d_out, int out_size, void* d_ws, size_t ws_size,
                              hipStream_t stream) {
    const float* obs   = (const float*)d_in[0];
    const float* means = (const float*)d_in[1];
    const float* stds  = (const float*)d_in[2];
    const float* noise = (const float*)d_in[3];
    float* out = (float*)d_out;

    float* dists = (float*)d_ws;
    int*   eidx  = (int*)(dists + PN);
    float* ew    = (float*)(eidx + KPAD);
    float* scal  = ew + KPAD;

    dtw_kernel   <<<PN / 32, 32, 0, stream>>>(obs, dists);
    select_kernel<<<1, 1024, 0, stream>>>(dists, eidx, ew, scal);
    cem_reduce_kernel<<<TN, 32, 0, stream>>>(means, stds, noise, eidx, ew, scal, out);
}
